// W8A16LinearLayer_84842783965258
// MI455X (gfx1250) — matrix operand order repaired and checked by reading, not compiled, not run on hardware
//
#include <hip/hip_runtime.h>

typedef __attribute__((ext_vector_type(16))) _Float16 v16h;
typedef __attribute__((ext_vector_type(8)))  _Float16 v8h;
typedef __attribute__((ext_vector_type(2)))  _Float16 v2h;
typedef __attribute__((ext_vector_type(8)))  float    v8f;
typedef __attribute__((ext_vector_type(4)))  float    v4f;
typedef __attribute__((ext_vector_type(4)))  int      v4i;

#define K_DIM  4096
#define N_DIM  11008
#define M_DIM  128
#define KSPLIT 2
#define KH     (K_DIM / KSPLIT)   // 2048 per z-block
#define BLK_M  64                 // rows per block; each wave covers all of them
#define BLK_N  256                // 8 waves x 32 columns
#define KC     64                 // K-chunk staged in LDS
#define LDSW   72                 // halves per row (64 + 8 pad, 144B, 16B aligned)
#define LDSBUF (BLK_M * LDSW)     // halves per LDS buffer

#define AS1 __attribute__((address_space(1)))
#define AS3 __attribute__((address_space(3)))

// exact int8 -> f16: (x^0x80)|0x6400 == f16(1152 + x); subtract 1152 -> x.
__device__ __forceinline__ v16h cvt_w16(v4i wq) {
    const unsigned int magic = 0x64646464u;
    v16h b;
#pragma unroll
    for (int j = 0; j < 4; ++j) {
        unsigned int t  = ((unsigned int)wq[j]) ^ 0x80808080u;
        unsigned int lo = __builtin_amdgcn_perm(magic, t, 0x05010400u);
        unsigned int hi = __builtin_amdgcn_perm(magic, t, 0x04030402u);
        v2h p0 = __builtin_bit_cast(v2h, lo) + __builtin_bit_cast(v2h, 0xE480E480u);
        v2h p1 = __builtin_bit_cast(v2h, hi) + __builtin_bit_cast(v2h, 0xE480E480u);
        b[4 * j + 0] = p0[0];
        b[4 * j + 1] = p0[1];
        b[4 * j + 2] = p1[0];
        b[4 * j + 3] = p1[1];
    }
    return b;
}

__device__ __forceinline__ void async_cp16(const _Float16* g, _Float16* l) {
    __builtin_amdgcn_global_load_async_to_lds_b128(
        (AS1 v4i*)g, (AS3 v4i*)l, 0, 0);
}

__global__ void __launch_bounds__(256) zero_out_kernel(float* __restrict__ Y) {
    const int i = blockIdx.x * blockDim.x + threadIdx.x;
    v4f z = {0.f, 0.f, 0.f, 0.f};
    ((v4f*)Y)[i] = z;
}

__global__ void __launch_bounds__(256)
cvt_x_kernel(const float* __restrict__ X, _Float16* __restrict__ Xh) {
    const int i = (blockIdx.x * blockDim.x + threadIdx.x) * 8;
    v4f a = *(const v4f*)(X + i);
    v4f b = *(const v4f*)(X + i + 4);
    v8h h;
#pragma unroll
    for (int j = 0; j < 4; ++j) { h[j] = (_Float16)a[j]; h[4 + j] = (_Float16)b[j]; }
    *(v8h*)(Xh + i) = h;
}

// ---------------- main GEMM: async A staging (f16 prepass in d_ws) -------------
__global__ void __launch_bounds__(256)
w8a16_wmma_async_kernel(const _Float16* __restrict__ Xh,
                        const signed char* __restrict__ W,
                        const float* __restrict__ scales,
                        const float* __restrict__ bias,
                        float* __restrict__ Y) {
    __shared__ _Float16 Ash[2 * LDSBUF];

    const int tid  = threadIdx.x;
    const int wave = tid >> 5;
    const int lane = tid & 31;
    const int half = lane >> 4;
    const int l16  = lane & 15;

    const int m_blk = blockIdx.y * BLK_M;
    const int n_wav = blockIdx.x * BLK_N + wave * 32;
    const int kz    = blockIdx.z * KH;

    v8f acc[4][2];
#pragma unroll
    for (int mt = 0; mt < 4; ++mt)
#pragma unroll
        for (int t = 0; t < 2; ++t)
#pragma unroll
            for (int i = 0; i < 8; ++i) acc[mt][t][i] = 0.0f;

    // A staging: 256 threads x 2 async B128 = 64x64 f16 chunk
    const int fill_row = tid >> 2;        // 0..63
    const int fill_k   = (tid & 3) * 16;  // 0,16,32,48
    const _Float16* Xg = Xh + (size_t)(m_blk + fill_row) * K_DIM + kz + fill_k;
    _Float16*       Al = Ash + fill_row * LDSW + fill_k;

    const signed char* Wp0 = W + (size_t)(n_wav + 0 * 16 + l16) * K_DIM + kz + half * 16;
    const signed char* Wp1 = W + (size_t)(n_wav + 1 * 16 + l16) * K_DIM + kz + half * 16;

    // prologue: chunk 0 -> buffer 0
    async_cp16(Xg, Al);
    async_cp16(Xg + 8, Al + 8);

    const int nchunks = KH / KC;          // 32
    for (int c = 0; c < nchunks; ++c) {
        const int k0 = c * KC;
        __syncthreads();   // all waves done reading buf[(c+1)&1] (chunk c-1 compute)
        if (c + 1 < nchunks) {
            _Float16* Aln = Ash + ((c + 1) & 1) * LDSBUF + fill_row * LDSW + fill_k;
            const _Float16* Xn = Xg + k0 + KC;
            async_cp16(Xn, Aln);
            async_cp16(Xn + 8, Aln + 8);
            // chunk c's 2 copies complete (in-order); chunk c+1 stays in flight
            asm volatile("s_wait_asynccnt 0x2" ::: "memory");
        } else {
            asm volatile("s_wait_asynccnt 0x0" ::: "memory");
        }
        __syncthreads();   // chunk c visible block-wide
        const _Float16* Abase = Ash + (c & 1) * LDSBUF;

#pragma unroll
        for (int ks = 0; ks < KC; ks += 32) {
            v4i wq0 = *(const v4i*)(Wp0 + k0 + ks);
            v4i wq1 = *(const v4i*)(Wp1 + k0 + ks);
            v16h b0 = cvt_w16(wq0);
            v16h b1 = cvt_w16(wq1);
#pragma unroll
            for (int mt = 0; mt < 4; ++mt) {
                const _Float16* ab = Abase + (mt * 16 + l16) * LDSW + ks + half * 16;
                v8h a_lo = *(const v8h*)(ab);
                v8h a_hi = *(const v8h*)(ab + 8);
                v16h a;
#pragma unroll
                for (int i = 0; i < 8; ++i) { a[i] = a_lo[i]; a[8 + i] = a_hi[i]; }
                acc[mt][0] = __builtin_amdgcn_wmma_f32_16x16x32_f16(
                    false, a, false, b0, (short)0, acc[mt][0], false, false);
                acc[mt][1] = __builtin_amdgcn_wmma_f32_16x16x32_f16(
                    false, a, false, b1, (short)0, acc[mt][1], false, false);
            }
        }
    }

    const bool add_bias = (blockIdx.z == 0);
#pragma unroll
    for (int t = 0; t < 2; ++t) {
        const int n   = n_wav + t * 16 + l16;
        const float s  = scales[n];
        const float bb = add_bias ? bias[n] : 0.0f;
#pragma unroll
        for (int mt = 0; mt < 4; ++mt) {
            const int m_base = m_blk + mt * 16 + half * 8;
#pragma unroll
            for (int r = 0; r < 8; ++r) {
                float v = acc[mt][t][r] * s + bb;
                unsafeAtomicAdd(&Y[(size_t)(m_base + r) * N_DIM + n], v);
            }
        }
    }
}

// ------------- fallback GEMM (no workspace): register-staged f32 -> f16 --------
__global__ void __launch_bounds__(256)
w8a16_wmma_sync_kernel(const float* __restrict__ X,
                       const signed char* __restrict__ W,
                       const float* __restrict__ scales,
                       const float* __restrict__ bias,
                       float* __restrict__ Y) {
    __shared__ _Float16 Ash[LDSBUF];

    const int tid  = threadIdx.x;
    const int wave = tid >> 5;
    const int lane = tid & 31;
    const int half = lane >> 4;
    const int l16  = lane & 15;

    const int m_blk = blockIdx.y * BLK_M;
    const int n_wav = blockIdx.x * BLK_N + wave * 32;
    const int kz    = blockIdx.z * KH;

    v8f acc[4][2];
#pragma unroll
    for (int mt = 0; mt < 4; ++mt)
#pragma unroll
        for (int t = 0; t < 2; ++t)
#pragma unroll
            for (int i = 0; i < 8; ++i) acc[mt][t][i] = 0.0f;

    const int fill_row = tid >> 2;
    const int fill_k   = (tid & 3) * 16;
    const float* Xp = X + (size_t)(m_blk + fill_row) * K_DIM + kz + fill_k;
    _Float16*   Aw  = Ash + fill_row * LDSW + fill_k;

    const signed char* Wp0 = W + (size_t)(n_wav + 0 * 16 + l16) * K_DIM + kz + half * 16;
    const signed char* Wp1 = W + (size_t)(n_wav + 1 * 16 + l16) * K_DIM + kz + half * 16;

    v4f x0 = *(const v4f*)(Xp + 0);
    v4f x1 = *(const v4f*)(Xp + 4);
    v4f x2 = *(const v4f*)(Xp + 8);
    v4f x3 = *(const v4f*)(Xp + 12);

    for (int k0 = 0; k0 < KH; k0 += KC) {
        v8h h0, h1;
#pragma unroll
        for (int i = 0; i < 4; ++i) {
            h0[i] = (_Float16)x0[i]; h0[4 + i] = (_Float16)x1[i];
            h1[i] = (_Float16)x2[i]; h1[4 + i] = (_Float16)x3[i];
        }
        *(v8h*)(Aw)     = h0;
        *(v8h*)(Aw + 8) = h1;
        __syncthreads();

        if (k0 + KC < KH) {
            x0 = *(const v4f*)(Xp + k0 + KC + 0);
            x1 = *(const v4f*)(Xp + k0 + KC + 4);
            x2 = *(const v4f*)(Xp + k0 + KC + 8);
            x3 = *(const v4f*)(Xp + k0 + KC + 12);
        }

#pragma unroll
        for (int ks = 0; ks < KC; ks += 32) {
            v4i wq0 = *(const v4i*)(Wp0 + k0 + ks);
            v4i wq1 = *(const v4i*)(Wp1 + k0 + ks);
            v16h b0 = cvt_w16(wq0);
            v16h b1 = cvt_w16(wq1);
#pragma unroll
            for (int mt = 0; mt < 4; ++mt) {
                const _Float16* ab = Ash + (mt * 16 + l16) * LDSW + ks + half * 16;
                v8h a_lo = *(const v8h*)(ab);
                v8h a_hi = *(const v8h*)(ab + 8);
                v16h a;
#pragma unroll
                for (int i = 0; i < 8; ++i) { a[i] = a_lo[i]; a[8 + i] = a_hi[i]; }
                acc[mt][0] = __builtin_amdgcn_wmma_f32_16x16x32_f16(
                    false, a, false, b0, (short)0, acc[mt][0], false, false);
                acc[mt][1] = __builtin_amdgcn_wmma_f32_16x16x32_f16(
                    false, a, false, b1, (short)0, acc[mt][1], false, false);
            }
        }
        __syncthreads();
    }

    const bool add_bias = (blockIdx.z == 0);
#pragma unroll
    for (int t = 0; t < 2; ++t) {
        const int n   = n_wav + t * 16 + l16;
        const float s  = scales[n];
        const float bb = add_bias ? bias[n] : 0.0f;
#pragma unroll
        for (int mt = 0; mt < 4; ++mt) {
            const int m_base = m_blk + mt * 16 + half * 8;
#pragma unroll
            for (int r = 0; r < 8; ++r) {
                float v = acc[mt][t][r] * s + bb;
                unsafeAtomicAdd(&Y[(size_t)(m_base + r) * N_DIM + n], v);
            }
        }
    }
}

extern "C" void kernel_launch(void* const* d_in, const int* in_sizes, int n_in,
                              void* d_out, int out_size, void* d_ws, size_t ws_size,
                              hipStream_t stream) {
    const float*       X  = (const float*)d_in[0];
    const signed char* W  = (const signed char*)d_in[1];
    const float*       Sc = (const float*)d_in[2];
    const float*       B  = (const float*)d_in[3];
    float*             Y  = (float*)d_out;

    const int n_v4 = (M_DIM * N_DIM) / 4;
    zero_out_kernel<<<n_v4 / 256, 256, 0, stream>>>(Y);

    dim3 grid(N_DIM / BLK_N, M_DIM / BLK_M, KSPLIT);   // (43, 2, 2)

    const size_t need = (size_t)M_DIM * K_DIM * sizeof(_Float16);   // 1 MB
    if (ws_size >= need) {
        _Float16* Xh = (_Float16*)d_ws;
        cvt_x_kernel<<<(M_DIM * K_DIM) / (256 * 8), 256, 0, stream>>>(X, Xh);
        w8a16_wmma_async_kernel<<<grid, dim3(256), 0, stream>>>(Xh, W, Sc, B, Y);
    } else {
        w8a16_wmma_sync_kernel<<<grid, dim3(256), 0, stream>>>(X, W, Sc, B, Y);
    }
}
